// ARAttention_22127671509580
// MI455X (gfx1250) — matrix operand order repaired and checked by reading, not compiled, not run on hardware
//
#include <hip/hip_runtime.h>
#include <hip/hip_bf16.h>
#include <cstdint>

typedef __attribute__((ext_vector_type(16))) __bf16 v16bf;
typedef __attribute__((ext_vector_type(8)))  float  v8f;

typedef unsigned short u16;
typedef unsigned int   u32;

union Frag { v16bf v; u32 d[8]; u16 s[16]; };

__device__ inline v8f vzero() {
    v8f z = {0.f,0.f,0.f,0.f,0.f,0.f,0.f,0.f};
    return z;
}

__device__ inline v8f wmma_bf16(v16bf a, v16bf b, v8f c) {
    // D = A(16x32 bf16) * B(32x16 bf16) + C(16x16 f32)
    return __builtin_amdgcn_wmma_f32_16x16x32_bf16(false, a, false, b, (short)0, c, false, false);
}

__device__ inline u16 f2bf(float f) {
    union { float f; u32 u; } x; x.f = f;
    u32 u = x.u;
    u32 r = u + 0x7FFFu + ((u >> 16) & 1u);   // RNE
    return (u16)(r >> 16);
}

__device__ inline float bf2f(u16 s) {
    union { u32 u; float f; } x; x.u = ((u32)s) << 16;
    return x.f;
}

// Map (branch, batch, window, pixel-in-window) -> global pixel row m in [0,16384)
// branch 0: vertical windows (2 rows x 64 cols); branch 1: horizontal (64 rows x 2 cols)
__device__ inline int pix_row(int branch, int n, int widx, int j) {
    if (branch == 0) return n * 4096 + (2 * widx + (j >> 6)) * 64 + (j & 63);
    else             return n * 4096 + (j >> 1) * 64 + 2 * widx + (j & 1);
}

#define QKSCALE 0.0625f   // 256^-0.5

// ---------------------------------------------------------------------------
// Kernel 1: QKV projection.  A = x (NCHW -> pixels x 256), B = Wqkv (256x768).
// Stores bf16 qkv (q pre-scaled by QKSCALE) and f32 v for the lepe conv.
// Block: 128 threads (4 waves), tile 64(M) x 64(N), K-loop 8 x 32.
// ---------------------------------------------------------------------------
__global__ __launch_bounds__(128) void qkv_gemm_k(
    const float* __restrict__ x, const float* __restrict__ W,
    const float* __restrict__ b, u16* __restrict__ qkv, float* __restrict__ vf)
{
    __shared__ u16 a_lds[64][32];   // [m][k] bf16
    __shared__ u16 bT[64][32];      // [n][k] bf16 (transposed)

    const int tid  = threadIdx.x;
    const int lane = tid & 31, wave = tid >> 5;
    const int laneN = lane & 15, half = lane >> 4;
    const int m0 = blockIdx.x * 64;
    const int n0 = blockIdx.y * 64;
    const int nb  = m0 >> 12;
    const int hw0 = m0 & 4095;

    v8f acc[4];
    #pragma unroll
    for (int j = 0; j < 4; ++j) acc[j] = vzero();

    for (int kb = 0; kb < 256; kb += 32) {
        {   // stage A: column k is 64 contiguous floats of x
            int k = tid >> 2, r0 = (tid & 3) * 16;
            const float* src = x + (size_t)(nb * 256 + kb + k) * 4096 + hw0 + r0;
            #pragma unroll
            for (int i = 0; i < 16; ++i) a_lds[r0 + i][k] = f2bf(src[i]);
        }
        {   // stage B transposed
            int k = tid >> 2, j0 = (tid & 3) * 16;
            const float* src = W + (size_t)(kb + k) * 768 + n0 + j0;
            #pragma unroll
            for (int i = 0; i < 16; ++i) bT[j0 + i][k] = f2bf(src[i]);
        }
        __syncthreads();

        Frag a;
        const int am = wave * 16 + laneN;
        #pragma unroll
        for (int p = 0; p < 8; ++p) {
            int k0 = ((p >> 2) << 4) + (half << 3) + ((p & 3) << 1);
            a.d[p] = *(const u32*)&a_lds[am][k0];
        }
        #pragma unroll
        for (int jt = 0; jt < 4; ++jt) {
            Frag bb;
            #pragma unroll
            for (int p = 0; p < 8; ++p) {
                int nn = jt * 16 + laneN;
                int kk = ((p >> 2) << 4) + (half << 3) + ((p & 3) << 1);
                bb.d[p] = *(const u32*)&bT[nn][kk];
            }
            acc[jt] = wmma_bf16(a.v, bb.v, acc[jt]);
        }
        __syncthreads();
    }

    #pragma unroll
    for (int jt = 0; jt < 4; ++jt) {
        #pragma unroll
        for (int r = 0; r < 8; ++r) {
            int m = m0 + wave * 16 + r + 8 * half;
            int c = n0 + jt * 16 + laneN;
            float v = acc[jt][r] + b[c];
            if (c < 256) v *= QKSCALE;          // fold attention scale into q
            qkv[(size_t)m * 768 + c] = f2bf(v);
            if (c >= 512) vf[(size_t)m * 256 + (c - 512)] = v;
        }
    }
}

// ---------------------------------------------------------------------------
// Kernel 2: per-window means of q (scaled) and k, both branches.
// Grid: 256 blocks = branch(2) x n(4) x p(32); 256 threads = channels.
// ---------------------------------------------------------------------------
__global__ __launch_bounds__(256) void win_means_k(
    const u16* __restrict__ qkv, float* __restrict__ wm)
{
    int bId = blockIdx.x;
    int p = bId & 31, n = (bId >> 5) & 3, branch = bId >> 7;
    int c = threadIdx.x;
    float aq = 0.f, ak = 0.f;
    for (int j = 0; j < 128; ++j) {
        int row = pix_row(branch, n, p, j);
        size_t base = (size_t)row * 768;
        aq += bf2f(qkv[base + c]);
        ak += bf2f(qkv[base + 256 + c]);
    }
    size_t o = ((size_t)branch * 128 + n * 32 + p) * 512;
    wm[o + c]       = aq * (1.f / 128.f);
    wm[o + 256 + c] = ak * (1.f / 128.f);
}

// ---------------------------------------------------------------------------
// Kernel 3: 32x32 routing logits + top-2 per row (ties -> lower index first).
// Grid: 8 blocks = branch(2) x n(4); 32 threads = query windows.
// ---------------------------------------------------------------------------
__global__ __launch_bounds__(32) void route_top2_k(
    const float* __restrict__ wm, int* __restrict__ ridx)
{
    int bId = blockIdx.x;
    int branch = bId >> 2, n = bId & 3;
    int p = threadIdx.x;
    const float* qrow = wm + ((size_t)branch * 128 + n * 32 + p) * 512;
    float v1 = -3.0e38f, v2 = -3.0e38f;
    int i1 = 0, i2 = 0;
    for (int q2 = 0; q2 < 32; ++q2) {
        const float* krow = wm + ((size_t)branch * 128 + n * 32 + q2) * 512 + 256;
        float d = 0.f;
        for (int c = 0; c < 256; ++c) d += qrow[c] * krow[c];
        if (d > v1)      { v2 = v1; i2 = i1; v1 = d; i1 = q2; }
        else if (d > v2) { v2 = d;  i2 = q2; }
    }
    int base = (branch * 128 + n * 32 + p) * 2;
    ridx[base] = i1;
    ridx[base + 1] = i2;
}

// ---------------------------------------------------------------------------
// Kernel 4: gathered-window attention. One wave per (branch,n,window,head).
// K staged [key][dim], V staged transposed [dim][key] in LDS; Q/A-frags read
// directly from global (bf16 pairs are contiguous -> dword loads).
// ---------------------------------------------------------------------------
__global__ __launch_bounds__(32) void attn_k(
    const u16* __restrict__ qkv, const int* __restrict__ ridx,
    float* __restrict__ obuf)
{
    __shared__ u16 k_lds[256][32];   // 16 KB
    __shared__ u16 vT[32][256];      // 16 KB
    __shared__ u16 p_lds[16][256];   //  8 KB

    const int bId = blockIdx.x;
    const int head = bId & 7, p = (bId >> 3) & 31, n = (bId >> 8) & 3, branch = bId >> 10;
    const int lane = threadIdx.x, laneN = lane & 15, half = lane >> 4;

    const int rbase = (branch * 128 + n * 32 + p) * 2;
    const int w0 = ridx[rbase], w1 = ridx[rbase + 1];

    // ---- stage K_sel and V_sel^T for this head ----
    for (int r = 0; r < 8; ++r) {
        int key = r * 32 + lane;
        int widx = (key < 128) ? w0 : w1;
        int row = pix_row(branch, n, widx, key & 127);
        const u32* ksrc = (const u32*)(qkv + (size_t)row * 768 + 256 + head * 32);
        const u32* vsrc = (const u32*)(qkv + (size_t)row * 768 + 512 + head * 32);
        u32* kd = (u32*)&k_lds[key][0];
        #pragma unroll
        for (int i = 0; i < 16; ++i) kd[i] = ksrc[i];
        #pragma unroll
        for (int i = 0; i < 16; ++i) {
            u32 u = vsrc[i];
            vT[2 * i][key]     = (u16)u;
            vT[2 * i + 1][key] = (u16)(u >> 16);
        }
    }
    __syncthreads();

    for (int mt = 0; mt < 8; ++mt) {
        // ---- Q A-fragment straight from global ----
        int qi = mt * 16 + laneN;
        int qrow = pix_row(branch, n, p, qi);
        Frag qa;
        #pragma unroll
        for (int pp = 0; pp < 8; ++pp) {
            int k0 = ((pp >> 2) << 4) + (half << 3) + ((pp & 3) << 1);
            qa.d[pp] = *(const u32*)(qkv + (size_t)qrow * 768 + head * 32 + k0);
        }

        // ---- scores: S = Q(16x32) * K^T, 16 key tiles ----
        v8f sc[16];
        #pragma unroll
        for (int nt = 0; nt < 16; ++nt) {
            Frag kb;
            #pragma unroll
            for (int pp = 0; pp < 8; ++pp) {
                int key = nt * 16 + laneN;
                int dd  = ((pp >> 2) << 4) + (half << 3) + ((pp & 3) << 1);
                kb.d[pp] = *(const u32*)&k_lds[key][dd];
            }
            sc[nt] = wmma_bf16(qa.v, kb.v, vzero());
        }

        // ---- softmax over 256 keys (row = r + 8*half) ----
        float rinv[8];
        #pragma unroll
        for (int r = 0; r < 8; ++r) {
            float mx = -3.0e38f;
            #pragma unroll
            for (int nt = 0; nt < 16; ++nt) mx = fmaxf(mx, sc[nt][r]);
            #pragma unroll
            for (int sh = 1; sh < 16; sh <<= 1) mx = fmaxf(mx, __shfl_xor(mx, sh, 32));
            float s = 0.f;
            #pragma unroll
            for (int nt = 0; nt < 16; ++nt) {
                float e = __expf(sc[nt][r] - mx);
                s += e;
                p_lds[r + 8 * half][nt * 16 + laneN] = f2bf(e);
            }
            #pragma unroll
            for (int sh = 1; sh < 16; sh <<= 1) s += __shfl_xor(s, sh, 32);
            rinv[r] = 1.f / s;
        }
        __syncthreads();

        // ---- O = P(16x256) * V(256x32) ----
        #pragma unroll
        for (int nt2 = 0; nt2 < 2; ++nt2) {
            v8f oc = vzero();
            #pragma unroll
            for (int ks = 0; ks < 8; ++ks) {
                Frag pa, vb;
                #pragma unroll
                for (int pp = 0; pp < 8; ++pp) {
                    int k0 = ((pp >> 2) << 4) + (half << 3) + ((pp & 3) << 1);
                    pa.d[pp] = *(const u32*)&p_lds[laneN][ks * 32 + k0];
                    int keyr = ks * 32 + ((pp >> 2) << 4) + (half << 3) + ((pp & 3) << 1);
                    vb.d[pp] = *(const u32*)&vT[nt2 * 16 + laneN][keyr];
                }
                oc = wmma_bf16(pa.v, vb.v, oc);
            }
            #pragma unroll
            for (int r = 0; r < 8; ++r) {
                float v = oc[r] * rinv[r];
                int qg = mt * 16 + r + 8 * half;
                int row = pix_row(branch, n, p, qg);
                obuf[((size_t)branch * 16384 + row) * 256 + head * 32 + nt2 * 16 + laneN] = v;
            }
        }
        __syncthreads();
    }
}

// ---------------------------------------------------------------------------
// Kernel 5: lepe depthwise 3x3 conv on v + combine both attention branches.
// simg = attn_v + attn_h + 2*(conv(v)+lepe_b), stored bf16 for final GEMM.
// Grid: 16384 pixels; 256 threads = channels.
// ---------------------------------------------------------------------------
__global__ __launch_bounds__(256) void lepe_combine_k(
    const float* __restrict__ obuf, const float* __restrict__ vf,
    const float* __restrict__ lepe_w, const float* __restrict__ lepe_b,
    u16* __restrict__ simg)
{
    int m = blockIdx.x;
    int n = m >> 12, hw = m & 4095, h = hw >> 6, w = hw & 63;
    int c = threadIdx.x;
    float acc = lepe_b[c];
    #pragma unroll
    for (int dh = 0; dh < 3; ++dh) {
        int hh = h + dh - 1;
        if (hh < 0 || hh > 63) continue;
        #pragma unroll
        for (int dw = 0; dw < 3; ++dw) {
            int ww = w + dw - 1;
            if (ww < 0 || ww > 63) continue;
            acc += vf[((size_t)n * 4096 + hh * 64 + ww) * 256 + c]
                 * lepe_w[(dh * 3 + dw) * 256 + c];
        }
    }
    float s = obuf[(size_t)m * 256 + c] + obuf[((size_t)16384 + m) * 256 + c] + 2.f * acc;
    simg[(size_t)m * 256 + c] = f2bf(s);
}

// ---------------------------------------------------------------------------
// Kernel 6: output projection  out = simg @ Wo + 2*bo, scatter to NCHW f32.
// ---------------------------------------------------------------------------
__global__ __launch_bounds__(128) void out_gemm_k(
    const u16* __restrict__ simg, const float* __restrict__ Wo,
    const float* __restrict__ bo, float* __restrict__ out)
{
    __shared__ u16 a_lds[64][32];
    __shared__ u16 bT[64][32];

    const int tid  = threadIdx.x;
    const int lane = tid & 31, wave = tid >> 5;
    const int laneN = lane & 15, half = lane >> 4;
    const int m0 = blockIdx.x * 64;
    const int n0 = blockIdx.y * 64;

    v8f acc[4];
    #pragma unroll
    for (int j = 0; j < 4; ++j) acc[j] = vzero();

    for (int kb = 0; kb < 256; kb += 32) {
        {   // stage A (already bf16, row-major)
            int r = tid >> 1, k0 = (tid & 1) * 16;
            const u16* src = simg + (size_t)(m0 + r) * 256 + kb + k0;
            #pragma unroll
            for (int i = 0; i < 16; ++i) a_lds[r][k0 + i] = src[i];
        }
        {   // stage B transposed
            int k = tid >> 2, j0 = (tid & 3) * 16;
            const float* src = Wo + (size_t)(kb + k) * 256 + n0 + j0;
            #pragma unroll
            for (int i = 0; i < 16; ++i) bT[j0 + i][k] = f2bf(src[i]);
        }
        __syncthreads();

        Frag a;
        const int am = wave * 16 + laneN;
        #pragma unroll
        for (int p = 0; p < 8; ++p) {
            int k0 = ((p >> 2) << 4) + (half << 3) + ((p & 3) << 1);
            a.d[p] = *(const u32*)&a_lds[am][k0];
        }
        #pragma unroll
        for (int jt = 0; jt < 4; ++jt) {
            Frag bb;
            #pragma unroll
            for (int p = 0; p < 8; ++p) {
                int nn = jt * 16 + laneN;
                int kk = ((p >> 2) << 4) + (half << 3) + ((p & 3) << 1);
                bb.d[p] = *(const u32*)&bT[nn][kk];
            }
            acc[jt] = wmma_bf16(a.v, bb.v, acc[jt]);
        }
        __syncthreads();
    }

    #pragma unroll
    for (int jt = 0; jt < 4; ++jt) {
        #pragma unroll
        for (int r = 0; r < 8; ++r) {
            int m = m0 + wave * 16 + r + 8 * half;
            int c = n0 + jt * 16 + laneN;
            float v = acc[jt][r] + 2.f * bo[c];
            out[((size_t)(m >> 12) * 256 + c) * 4096 + (m & 4095)] = v;   // NCHW
        }
    }
}

// ---------------------------------------------------------------------------
extern "C" void kernel_launch(void* const* d_in, const int* in_sizes, int n_in,
                              void* d_out, int out_size, void* d_ws, size_t ws_size,
                              hipStream_t stream)
{
    (void)in_sizes; (void)n_in; (void)out_size; (void)ws_size;

    const float* x      = (const float*)d_in[0];   // (4,256,64,64)
    const float* Wqkv   = (const float*)d_in[1];   // (256,768)
    const float* bqkv   = (const float*)d_in[2];   // (768,)
    const float* Wo     = (const float*)d_in[3];   // (256,256)
    const float* bo     = (const float*)d_in[4];   // (256,)
    const float* lepe_w = (const float*)d_in[5];   // (3,3,1,256)
    const float* lepe_b = (const float*)d_in[6];   // (256,)
    float* out = (float*)d_out;                    // (4,256,64,64)

    // workspace layout (bytes)
    char* w = (char*)d_ws;
    u16*   qkv  = (u16*)  (w);                       // 16384*768*2  = 25165824
    float* vf   = (float*)(w + 25165824);            // 16384*256*4  = 16777216
    float* obuf = (float*)(w + 41943040);            // 2*16384*256*4 = 33554432
    u16*   simg = (u16*)  (w + 75497472);            // 16384*256*2  = 8388608
    float* wm   = (float*)(w + 83886080);            // 2*128*512*4  = 524288
    int*   ridx = (int*)  (w + 84410368);            // 2*128*2*4    = 2048

    qkv_gemm_k   <<<dim3(256, 12), 128, 0, stream>>>(x, Wqkv, bqkv, qkv, vf);
    win_means_k  <<<256, 256, 0, stream>>>(qkv, wm);
    route_top2_k <<<8, 32, 0, stream>>>(wm, ridx);
    attn_k       <<<2048, 32, 0, stream>>>(qkv, ridx, obuf);
    lepe_combine_k<<<16384, 256, 0, stream>>>(obuf, vf, lepe_w, lepe_b, simg);
    out_gemm_k   <<<dim3(256, 4), 128, 0, stream>>>(simg, Wo, bo, out);
}
